// MRTransformer_69904887710002
// MI455X (gfx1250) — compile-verified, matrix operand order checked by reading, not run on hardware
//
#include <hip/hip_runtime.h>
#include <hip/hip_bf16.h>

// ---------------- problem constants (from reference) ----------------
#define B_    4
#define S_    2000
#define N_    (B_ * S_)        // 8000
#define E_    32000
#define H_TR  4
#define D_TR  64
#define DH_   (D_TR / H_TR)    // 16
#define FF_   2048
#define TOUT_ 16
#define HID_  512
#define HEADS_ 6
#define HC_   512
#define GHID_ (HEADS_ * HC_)   // 3072
#define DEFORM_IN_ 7
#define GIN_  (2 + HID_)       // 514
#define ALLF_ (DEFORM_IN_ - 2 + TOUT_)  // 21
#define NUM_LOOP_ 3
#define NE_   (E_ + N_)        // edges + self loops = 40000

typedef __attribute__((ext_vector_type(16))) _Float16     v16h;
typedef __attribute__((ext_vector_type(8)))  float        v8f;
typedef __attribute__((ext_vector_type(4)))  unsigned int u32x4;
typedef __attribute__((ext_vector_type(8)))  int          i32x8;
typedef __attribute__((ext_vector_type(4)))  int          i32x4;

#if defined(__has_builtin)
#if __has_builtin(__builtin_amdgcn_tensor_load_to_lds) && \
    __has_builtin(__builtin_amdgcn_s_wait_tensorcnt)
#define HAVE_TDM 1
#endif
#endif
#ifndef HAVE_TDM
#define HAVE_TDM 0
#endif

// ---------------- small device helpers ----------------
__device__ __forceinline__ float selu_f(float x) {
    const float a  = 1.6732632423543772848f;
    const float sc = 1.0507009873554804934f;
    return sc * (x > 0.f ? x : a * (__expf(x) - 1.f));
}

// CDNA5 16-bit A/B fragment K index for element e (0..15), lane half g (0/1)
__device__ __forceinline__ int wmma_kmap(int e, int g) {
    return g * 8 + e + (e >= 8 ? 8 : 0);
}

// monotonic uint encoding of float for atomicMax-based segment_max
__device__ __forceinline__ unsigned fenc(float f) {
    unsigned u = __float_as_uint(f);
    return (u & 0x80000000u) ? ~u : (u | 0x80000000u);
}
__device__ __forceinline__ float fdec(unsigned k) {
    unsigned u = (k & 0x80000000u) ? (k & 0x7FFFFFFFu) : ~k;
    return __uint_as_float(u);
}

// ---------------- generic WMMA GEMM: C = act(A[M,K] * W[N,K]^T + bias) ----------------
// block = 256 threads = 8 waves (wave32). Each wave owns 16 M-rows and computes a
// 16x64 strip (4 accumulators). The shared 64(N) x 32(K) fp32 W-tile is staged into
// LDS once per k-step — by the Tensor Data Mover (2D D#, hardware zero-fill OOB)
// when available, else by a cooperative vector-load fallback.
// ACT: 0 = none, 1 = relu, 2 = selu
template <int ACT>
__global__ void k_gemm(const float* __restrict__ A, const float* __restrict__ W,
                       const float* __restrict__ bias, float* __restrict__ C,
                       int M, int N, int K) {
    __shared__ float ldsB[64 * 32];      // [n_local][k_local], 8 KB

    const int lane = threadIdx.x & 31;
    const int wave = threadIdx.x >> 5;
    const int g  = lane >> 4;            // lane half
    const int li = lane & 15;
    const int mBase = blockIdx.y * 128 + wave * 16;
    const int nBase = blockIdx.x * 64;
    const int aRow  = mBase + li;

    v8f acc0 = {}, acc1 = {}, acc2 = {}, acc3 = {};

    for (int k0 = 0; k0 < K; k0 += 32) {
        __syncthreads();                 // previous tile fully consumed
#if HAVE_TDM
        if (threadIdx.x < 32) {          // wave 0 drives the TDM
            const unsigned long long gaddr =
                (unsigned long long)(uintptr_t)W +
                ((unsigned long long)nBase * (unsigned)K + (unsigned)k0) * 4ull;
            // low 32 bits of the flat shared-pointer = LDS byte offset
            const unsigned ldsOff = (unsigned)(uintptr_t)&ldsB[0];
            const int tdim0 = K - k0;    // remaining K (HW zero-fills beyond)
            const int tdim1 = N - nBase; // remaining N rows
            u32x4 g0;
            g0[0] = 1u;                                          // count=1, user D#
            g0[1] = ldsOff;                                      // lds_addr
            g0[2] = (unsigned)(gaddr & 0xFFFFFFFFull);           // global_addr lo
            g0[3] = (unsigned)((gaddr >> 32) & 0x1FFFFFFull)     // global_addr hi
                    | (2u << 30);                                // type = 2 (image)
            i32x8 g1;
            g1[0] = (2 << 16);                                   // data_size = 4B
            g1[1] = (tdim0 & 0xFFFF) << 16;                      // tensor_dim0 lo16
            g1[2] = ((tdim0 >> 16) & 0xFFFF) | ((tdim1 & 0xFFFF) << 16);
            g1[3] = ((tdim1 >> 16) & 0xFFFF) | (32 << 16);       // tile_dim0 = 32
            g1[4] = 64;                                          // tile_dim1 = 64
            g1[5] = K;                                           // dim0 stride lo32
            g1[6] = 0;
            g1[7] = 0;
            i32x4 z4 = {0, 0, 0, 0};
            i32x8 z8 = {0, 0, 0, 0, 0, 0, 0, 0};
            __builtin_amdgcn_tensor_load_to_lds(g0, g1, z4, z4, z8, 0);
            __builtin_amdgcn_s_wait_tensorcnt(0);
        }
#else
        // cooperative fallback: 2048 floats, 8 per thread
        for (int i = threadIdx.x; i < 64 * 32; i += 256) {
            const int nl = i >> 5, kl = i & 31;
            const int n = nBase + nl, kk = k0 + kl;
            ldsB[i] = (n < N && kk < K) ? W[(long)n * K + kk] : 0.f;
        }
#endif
        __syncthreads();

        // A fragment from global (each row read by exactly one wave)
        v16h af;
#pragma unroll
        for (int e = 0; e < 16; ++e) {
            const int kk = k0 + wmma_kmap(e, g);
            af[e] = (_Float16)((aRow < M && kk < K) ? A[(long)aRow * K + kk] : 0.f);
        }
        // 4 B fragments from LDS, 4 WMMAs reusing af
#pragma unroll
        for (int t = 0; t < 4; ++t) {
            v16h bf;
#pragma unroll
            for (int e = 0; e < 16; ++e) {
                const int kl = wmma_kmap(e, g);
                bf[e] = (_Float16)ldsB[(t * 16 + li) * 32 + kl];
            }
            v8f& acc = (t == 0) ? acc0 : (t == 1) ? acc1 : (t == 2) ? acc2 : acc3;
            acc = __builtin_amdgcn_wmma_f32_16x16x32_f16(
                false, af, false, bf, (short)0, acc, false, false);
        }
    }

#pragma unroll
    for (int t = 0; t < 4; ++t) {
        const v8f acc = (t == 0) ? acc0 : (t == 1) ? acc1 : (t == 2) ? acc2 : acc3;
        const int n = nBase + t * 16 + li;
#pragma unroll
        for (int r = 0; r < 8; ++r) {
            const int m = mBase + r + g * 8;
            if (m < M && n < N) {
                float v = acc[r] + bias[n];
                if (ACT == 1) v = fmaxf(v, 0.f);
                if (ACT == 2) v = selu_f(v);
                C[(long)m * N + n] = v;
            }
        }
    }
}

// ---------------- embed: h[n,d] = mesh[n,:4] @ embed_w[d,:4] + embed_b[d] ----------------
__global__ void k_embed(const float* __restrict__ mesh, const float* __restrict__ w,
                        const float* __restrict__ b, float* __restrict__ h) {
    long i = (long)blockIdx.x * blockDim.x + threadIdx.x;
    if (i >= (long)N_ * D_TR) return;
    int n = (int)(i >> 6), d = (int)(i & 63);
    float s = b[d];
#pragma unroll
    for (int j = 0; j < 4; ++j) s += mesh[n * 4 + j] * w[d * 4 + j];
    h[i] = s;
}

// ---------------- flash attention, one wave per (b, h, 16-query tile) ----------------
__global__ void k_attn(const float* __restrict__ qkv, float* __restrict__ o) {
    const int qt = blockIdx.x;
    const int h  = blockIdx.y;
    const int b  = blockIdx.z;
    const int lane = threadIdx.x & 31;
    const int g  = lane >> 4, li = lane & 15;
    __shared__ float pLds[16 * 16];

    const float scale = 0.25f;           // 1/sqrt(16)

    v16h qf;
    const long qRow = (long)b * S_ + qt * 16 + li;
#pragma unroll
    for (int e = 0; e < 16; ++e) {
        const int kk = wmma_kmap(e, g);
        float v = (kk < DH_) ? qkv[qRow * 192 + h * DH_ + kk] * scale : 0.f;
        qf[e] = (_Float16)v;
    }

    v8f oacc = {};
    float runm[8], runl[8];
#pragma unroll
    for (int r = 0; r < 8; ++r) { runm[r] = -1e30f; runl[r] = 0.f; }

    for (int kt = 0; kt < S_ / 16; ++kt) {
        v16h kf;
        const long kRow = (long)b * S_ + kt * 16 + li;
#pragma unroll
        for (int e = 0; e < 16; ++e) {
            const int kk = wmma_kmap(e, g);
            kf[e] = (kk < DH_) ? (_Float16)qkv[kRow * 192 + 64 + h * DH_ + kk]
                               : (_Float16)0.f;
        }
        v8f s = {};
        s = __builtin_amdgcn_wmma_f32_16x16x32_f16(
            false, qf, false, kf, (short)0, s, false, false);

#pragma unroll
        for (int r = 0; r < 8; ++r) {
            float x = s[r];
            for (int mk = 1; mk < 16; mk <<= 1) x = fmaxf(x, __shfl_xor(x, mk, 32));
            const float newm = fmaxf(runm[r], x);
            const float p = __expf(s[r] - newm);
            float rs = p;
            for (int mk = 1; mk < 16; mk <<= 1) rs += __shfl_xor(rs, mk, 32);
            const float f = __expf(runm[r] - newm);
            runl[r] = runl[r] * f + rs;
            runm[r] = newm;
            oacc[r] *= f;
            pLds[(r + 8 * g) * 16 + li] = p;
        }
        __syncthreads();

        v16h pf, vf;
#pragma unroll
        for (int e = 0; e < 16; ++e) {
            const int kk = wmma_kmap(e, g);
            pf[e] = (kk < 16) ? (_Float16)pLds[li * 16 + kk] : (_Float16)0.f;
            const long vRow = (long)b * S_ + kt * 16 + kk;
            vf[e] = (kk < 16) ? (_Float16)qkv[vRow * 192 + 128 + h * DH_ + li]
                              : (_Float16)0.f;
        }
        oacc = __builtin_amdgcn_wmma_f32_16x16x32_f16(
            false, pf, false, vf, (short)0, oacc, false, false);
        __syncthreads();
    }

#pragma unroll
    for (int r = 0; r < 8; ++r) {
        const int m = qt * 16 + r + 8 * g;
        o[((long)b * S_ + m) * 64 + h * DH_ + li] = oacc[r] / runl[r];
    }
}

// ---------------- residual + layernorm over D=64, thread-per-row ----------------
__global__ void k_add_ln(const float* __restrict__ a, const float* __restrict__ b,
                         const float* __restrict__ gamma, const float* __restrict__ beta,
                         float* __restrict__ out, int M) {
    int row = blockIdx.x * blockDim.x + threadIdx.x;
    if (row >= M) return;
    const float* pa = a + (long)row * 64;
    const float* pb = b + (long)row * 64;
    float mean = 0.f;
#pragma unroll
    for (int d = 0; d < 64; ++d) mean += pa[d] + pb[d];
    mean *= (1.f / 64.f);
    float var = 0.f;
#pragma unroll
    for (int d = 0; d < 64; ++d) {
        float t = pa[d] + pb[d] - mean;
        var += t * t;
    }
    var *= (1.f / 64.f);
    const float inv = rsqrtf(var + 1e-5f);
    float* po = out + (long)row * 64;
#pragma unroll
    for (int d = 0; d < 64; ++d)
        po[d] = (pa[d] + pb[d] - mean) * inv * gamma[d] + beta[d];
}

// ---------------- feats = concat(x[:,2:7], tout) -> [N,21] ----------------
__global__ void k_concat_feats(const float* __restrict__ x, const float* __restrict__ t,
                               float* __restrict__ feats) {
    long i = (long)blockIdx.x * blockDim.x + threadIdx.x;
    if (i >= (long)N_ * ALLF_) return;
    int n = (int)(i / ALLF_), c = (int)(i % ALLF_);
    feats[i] = (c < 5) ? x[n * DEFORM_IN_ + 2 + c] : t[(long)n * TOUT_ + (c - 5)];
}

// ---------------- coord init: coord = x[:, :2] ----------------
__global__ void k_init_coord(const float* __restrict__ x, float* __restrict__ coord) {
    long i = (long)blockIdx.x * blockDim.x + threadIdx.x;
    if (i >= (long)N_ * 2) return;
    int n = (int)(i >> 1), j = (int)(i & 1);
    coord[i] = x[n * DEFORM_IN_ + j];
}

// ---------------- inp = concat(coord, hidden) -> [N, 514] ----------------
__global__ void k_concat_inp(const float* __restrict__ coord, const float* __restrict__ hidden,
                             float* __restrict__ inp) {
    long i = (long)blockIdx.x * blockDim.x + threadIdx.x;
    if (i >= (long)N_ * GIN_) return;
    int n = (int)(i / GIN_), c = (int)(i % GIN_);
    inp[i] = (c < 2) ? coord[n * 2 + c] : hidden[(long)n * HID_ + (c - 2)];
}

// ---------------- GAT edge pass 1: logits + segment_max ----------------
__global__ void k_edge_logits(const float* __restrict__ xl, const float* __restrict__ xr,
                              const int* __restrict__ ei, const float* __restrict__ att,
                              float* __restrict__ logits, unsigned* __restrict__ mxenc) {
    const int e = blockIdx.x;
    const int head = threadIdx.x >> 5;
    const int lane = threadIdx.x & 31;
    int src, dst;
    if (e < E_) { src = ei[e]; dst = ei[E_ + e]; }
    else        { src = dst = e - E_; }
    const float* pl = xl + (long)src * GHID_ + head * HC_;
    const float* pr = xr + (long)dst * GHID_ + head * HC_;
    const float* pa = att + head * HC_;
    float s = 0.f;
    for (int c = lane; c < HC_; c += 32) {
        float v = pl[c] + pr[c];
        v = (v > 0.f) ? v : 0.2f * v;          // leaky_relu(., 0.2)
        s += v * pa[c];
    }
    for (int mk = 1; mk < 32; mk <<= 1) s += __shfl_xor(s, mk, 32);
    if (lane == 0) {
        logits[(long)e * HEADS_ + head] = s;
        atomicMax(&mxenc[(long)dst * HEADS_ + head], fenc(s));
    }
}

// ---------------- GAT edge pass 2: a = exp(logit - max); segment_sum denom ----------------
__global__ void k_edge_softmax(const float* __restrict__ logits, const unsigned* __restrict__ mxenc,
                               const int* __restrict__ ei, float* __restrict__ aexp,
                               float* __restrict__ denom) {
    long i = (long)blockIdx.x * blockDim.x + threadIdx.x;
    if (i >= (long)NE_ * HEADS_) return;
    int e = (int)(i / HEADS_), head = (int)(i % HEADS_);
    int dst = (e < E_) ? ei[E_ + e] : e - E_;
    float mx = fdec(mxenc[(long)dst * HEADS_ + head]);
    float a = __expf(logits[i] - mx);
    aexp[i] = a;
    atomicAdd(&denom[(long)dst * HEADS_ + head], a);
}

// ---------------- GAT edge pass 3: agg[dst] += xl[src] * a ----------------
__global__ void k_edge_scatter(const float* __restrict__ xl, const int* __restrict__ ei,
                               const float* __restrict__ aexp, const float* __restrict__ denom,
                               float* __restrict__ agg) {
    const int e = blockIdx.x;
    int src, dst;
    if (e < E_) { src = ei[e]; dst = ei[E_ + e]; }
    else        { src = dst = e - E_; }
    // pull the gathered row toward the WGP before the atomic storm
    __builtin_prefetch(&xl[(long)src * GHID_ + threadIdx.x], 0, 0);
    for (int hc = threadIdx.x; hc < GHID_; hc += blockDim.x) {
        const int head = hc >> 9;
        const float w = aexp[(long)e * HEADS_ + head] / denom[(long)dst * HEADS_ + head];
        atomicAdd(&agg[(long)dst * GHID_ + hc], xl[(long)src * GHID_ + hc] * w);
    }
}

// ---------------- GAT finish: hidden = selu(mean_h(agg) + bias) ----------------
__global__ void k_gat_finish(const float* __restrict__ agg, const float* __restrict__ bias,
                             float* __restrict__ hidden) {
    long i = (long)blockIdx.x * blockDim.x + threadIdx.x;
    if (i >= (long)N_ * HC_) return;
    int n = (int)(i >> 9), c = (int)(i & 511);
    float s = 0.f;
#pragma unroll
    for (int h = 0; h < HEADS_; ++h) s += agg[(long)n * GHID_ + h * HC_ + c];
    hidden[i] = selu_f(s * (1.f / HEADS_) + bias[c]);
}

// ---------------- host side ----------------
static inline char* ws_alloc(char*& p, size_t bytes) {
    char* r = p;
    p += (bytes + 255) & ~(size_t)255;
    return r;
}

static inline dim3 gemm_grid(int M, int N) {
    return dim3((N + 63) / 64, (M + 127) / 128, 1);
}

extern "C" void kernel_launch(void* const* d_in, const int* in_sizes, int n_in,
                              void* d_out, int out_size, void* d_ws, size_t ws_size,
                              hipStream_t stream) {
    (void)in_sizes; (void)n_in; (void)out_size; (void)ws_size;

    const float* mesh_feat  = (const float*)d_in[1];
    const float* x          = (const float*)d_in[2];
    const int*   edge_index = (const int*)  d_in[3];
    const float* embed_w    = (const float*)d_in[4];
    const float* embed_b    = (const float*)d_in[5];
    const float* in_proj_w  = (const float*)d_in[6];
    const float* in_proj_b  = (const float*)d_in[7];
    const float* out_proj_w = (const float*)d_in[8];
    const float* out_proj_b = (const float*)d_in[9];
    const float* ln1_g      = (const float*)d_in[10];
    const float* ln1_b      = (const float*)d_in[11];
    const float* ln2_g      = (const float*)d_in[12];
    const float* ln2_b      = (const float*)d_in[13];
    const float* ff1_w      = (const float*)d_in[14];
    const float* ff1_b      = (const float*)d_in[15];
    const float* ff2_w      = (const float*)d_in[16];
    const float* ff2_b      = (const float*)d_in[17];
    const float* tout_w     = (const float*)d_in[18];
    const float* tout_b     = (const float*)d_in[19];
    const float* lin_w      = (const float*)d_in[20];
    const float* lin_b      = (const float*)d_in[21];
    const float* gat_wl     = (const float*)d_in[22];
    const float* gat_bl     = (const float*)d_in[23];
    const float* gat_wr     = (const float*)d_in[24];
    const float* gat_br     = (const float*)d_in[25];
    const float* gat_att    = (const float*)d_in[26];
    const float* gat_bias   = (const float*)d_in[27];
    const float* coord1_w   = (const float*)d_in[28];
    const float* coord1_b   = (const float*)d_in[29];
    const float* coord2_w   = (const float*)d_in[30];
    const float* coord2_b   = (const float*)d_in[31];

    char* p = (char*)d_ws;
    float*    h      = (float*)   ws_alloc(p, (size_t)N_ * D_TR * 4);
    float*    qkv    = (float*)   ws_alloc(p, (size_t)N_ * 3 * D_TR * 4);
    float*    o      = (float*)   ws_alloc(p, (size_t)N_ * D_TR * 4);
    float*    tmp64  = (float*)   ws_alloc(p, (size_t)N_ * D_TR * 4);
    float*    h1     = (float*)   ws_alloc(p, (size_t)N_ * D_TR * 4);
    float*    ffi    = (float*)   ws_alloc(p, (size_t)N_ * FF_ * 4);
    float*    h2     = (float*)   ws_alloc(p, (size_t)N_ * D_TR * 4);
    float*    toutb  = (float*)   ws_alloc(p, (size_t)N_ * TOUT_ * 4);
    float*    feats  = (float*)   ws_alloc(p, (size_t)N_ * ALLF_ * 4);
    float*    hidden = (float*)   ws_alloc(p, (size_t)N_ * HID_ * 4);
    float*    inp    = (float*)   ws_alloc(p, (size_t)N_ * GIN_ * 4);
    float*    xl     = (float*)   ws_alloc(p, (size_t)N_ * GHID_ * 4);
    float*    xr     = (float*)   ws_alloc(p, (size_t)N_ * GHID_ * 4);
    float*    logits = (float*)   ws_alloc(p, (size_t)NE_ * HEADS_ * 4);
    float*    aexp   = (float*)   ws_alloc(p, (size_t)NE_ * HEADS_ * 4);
    unsigned* mxenc  = (unsigned*)ws_alloc(p, (size_t)N_ * HEADS_ * 4);
    float*    denom  = (float*)   ws_alloc(p, (size_t)N_ * HEADS_ * 4);
    float*    agg    = (float*)   ws_alloc(p, (size_t)N_ * GHID_ * 4);
    float*    c1     = (float*)   ws_alloc(p, (size_t)N_ * (HID_ / 2) * 4);
    float*    coord  = (float*)   ws_alloc(p, (size_t)N_ * 2 * 4);

    const int T = 256;
    auto cdiv = [](long a, long b) { return (int)((a + b - 1) / b); };

    // ---- transformer trunk ----
    k_embed<<<cdiv((long)N_ * D_TR, T), T, 0, stream>>>(mesh_feat, embed_w, embed_b, h);

    k_gemm<0><<<gemm_grid(N_, 3 * D_TR), T, 0, stream>>>(h, in_proj_w, in_proj_b, qkv,
                                                         N_, 3 * D_TR, D_TR);

    k_attn<<<dim3(S_ / 16, H_TR, B_), 32, 0, stream>>>(qkv, o);

    k_gemm<0><<<gemm_grid(N_, D_TR), T, 0, stream>>>(o, out_proj_w, out_proj_b, tmp64,
                                                     N_, D_TR, D_TR);
    k_add_ln<<<cdiv(N_, T), T, 0, stream>>>(h, tmp64, ln1_g, ln1_b, h1, N_);

    k_gemm<1><<<gemm_grid(N_, FF_), T, 0, stream>>>(h1, ff1_w, ff1_b, ffi, N_, FF_, D_TR);
    k_gemm<0><<<gemm_grid(N_, D_TR), T, 0, stream>>>(ffi, ff2_w, ff2_b, tmp64, N_, D_TR, FF_);
    k_add_ln<<<cdiv(N_, T), T, 0, stream>>>(h1, tmp64, ln2_g, ln2_b, h2, N_);

    k_gemm<0><<<gemm_grid(N_, TOUT_), T, 0, stream>>>(h2, tout_w, tout_b, toutb, N_, TOUT_, D_TR);

    k_concat_feats<<<cdiv((long)N_ * ALLF_, T), T, 0, stream>>>(x, toutb, feats);
    k_gemm<2><<<gemm_grid(N_, HID_), T, 0, stream>>>(feats, lin_w, lin_b, hidden, N_, HID_, ALLF_);
    k_init_coord<<<cdiv((long)N_ * 2, T), T, 0, stream>>>(x, coord);

    // ---- GAT loops ----
    for (int l = 0; l < NUM_LOOP_; ++l) {
        k_concat_inp<<<cdiv((long)N_ * GIN_, T), T, 0, stream>>>(coord, hidden, inp);

        k_gemm<0><<<gemm_grid(N_, GHID_), T, 0, stream>>>(inp, gat_wl, gat_bl, xl,
                                                          N_, GHID_, GIN_);
        k_gemm<0><<<gemm_grid(N_, GHID_), T, 0, stream>>>(inp, gat_wr, gat_br, xr,
                                                          N_, GHID_, GIN_);

        (void)hipMemsetAsync(mxenc, 0, (size_t)N_ * HEADS_ * 4, stream);
        (void)hipMemsetAsync(denom, 0, (size_t)N_ * HEADS_ * 4, stream);
        (void)hipMemsetAsync(agg,   0, (size_t)N_ * GHID_ * 4, stream);

        k_edge_logits<<<NE_, HEADS_ * 32, 0, stream>>>(xl, xr, edge_index, gat_att,
                                                       logits, mxenc);
        k_edge_softmax<<<cdiv((long)NE_ * HEADS_, T), T, 0, stream>>>(logits, mxenc,
                                                                      edge_index, aexp, denom);
        k_edge_scatter<<<NE_, T, 0, stream>>>(xl, edge_index, aexp, denom, agg);
        k_gat_finish<<<cdiv((long)N_ * HC_, T), T, 0, stream>>>(agg, gat_bias, hidden);

        k_gemm<2><<<gemm_grid(N_, HID_ / 2), T, 0, stream>>>(hidden, coord1_w, coord1_b, c1,
                                                             N_, HID_ / 2, HID_);
        k_gemm<0><<<gemm_grid(N_, 2), T, 0, stream>>>(c1, coord2_w, coord2_b, coord,
                                                      N_, 2, HID_ / 2);
    }

    (void)hipMemcpyAsync(d_out, coord, (size_t)N_ * 2 * 4, hipMemcpyDeviceToDevice, stream);
}
